// HashPINN_35665408426720
// MI455X (gfx1250) — matrix operand order repaired and checked by reading, not compiled, not run on hardware
//
#include <hip/hip_runtime.h>
#include <hip/hip_bf16.h>
#include <stdint.h>

// ---------------------------------------------------------------------------
// HashPINN fused kernel for gfx1250 (MI455X):
//   phase 1: per-thread Instant-NGP hash-grid encode -> f16 features in LDS
//   phase 2: per-wave WMMA MLP (32->64->64->64->1) via v_wmma_f32_16x16x32_f16
// 64 MB hash table is L2-resident (192 MB L2); MLP runs on the matrix pipes.
// Fragments are assembled with 16-byte ds_load_b128 accesses: A rows are
// contiguous per lane; weights are staged TRANSPOSED in LDS so B columns are
// contiguous per lane.
// ---------------------------------------------------------------------------

typedef __attribute__((ext_vector_type(16))) _Float16 v16h;
typedef __attribute__((ext_vector_type(8)))  _Float16 v8h;
typedef __attribute__((ext_vector_type(8)))  float    v8f;

#define NLEV     16
#define TBL_SZ   (1 << 19)          // hashmap entries per level
#define HSTRIDE  72                 // padded row stride (f16) for hidden tiles (144B, 16B-mult)

// Instant-NGP geometric progression, BASE_RES=16 .. MAX_RES=2048, 16 levels.
constexpr int RES[NLEV] = {16, 23, 31, 43, 59, 81, 112, 154,
                           213, 295, 407, 562, 777, 1073, 1483, 2048};

// ---------------------------------------------------------------------------
// WMMA fragment helpers; A/B use this file's contiguous K order (same for both, not the ISA table order).
// A (16x32 f16): lane%16 = M; elem e: K = 16*(lane>=16) + e
//   -> two contiguous 8xf16 (16B) runs per lane.
// B (32x16 f16): lane%16 = N; elem e: K = 16*(lane>=16) + e
//   -> contiguous in K; load from an [N][K] (transposed) LDS image.
// C/D (16x16 f32): VGPR r, lane l -> M = r + 8*(l>=16), N = l%16
// ---------------------------------------------------------------------------
__device__ __forceinline__ v16h load_A16x32(const _Float16* base, int stride, int lane) {
  const int m = lane & 15, half = lane >> 4;
  const _Float16* p = base + m * stride + (half << 4);
  const v8h lo = *reinterpret_cast<const v8h*>(p);        // K = 16*half + 0..7
  const v8h hi = *reinterpret_cast<const v8h*>(p + 8);    // K = 16*half + 8..15
  v16h a;
#pragma unroll
  for (int e = 0; e < 8; ++e) { a[e] = lo[e]; a[8 + e] = hi[e]; }
  return a;
}

// baseT points at Wt[n][k] image with row stride kstride (f16 units);
// koff selects the K-halves (0 or 32) for K=64 layers.
__device__ __forceinline__ v16h load_Bt32x16(const _Float16* baseT, int kstride,
                                             int ncol0, int koff, int lane) {
  const int n  = (lane & 15) + ncol0;
  const int kb = (lane >> 4) << 4;
  const _Float16* p = baseT + n * kstride + koff + kb;
  const v8h lo = *reinterpret_cast<const v8h*>(p);        // K = koff + kb + 0..7
  const v8h hi = *reinterpret_cast<const v8h*>(p + 8);    // K = koff + kb + 8..15
  v16h b;
#pragma unroll
  for (int e = 0; e < 8; ++e) { b[e] = lo[e]; b[8 + e] = hi[e]; }
  return b;
}

__device__ __forceinline__ void store_relu(_Float16* hrow, const v8f& acc,
                                           const float* bias, int ncol0, int lane) {
  const int n = (lane & 15) + ncol0, half = lane >> 4;
  const float bv = bias[n];
#pragma unroll
  for (int r = 0; r < 8; ++r) {
    float v = acc[r] + bv;
    v = v > 0.f ? v : 0.f;
    hrow[(r + half * 8) * HSTRIDE + n] = (_Float16)v;
  }
}

__device__ __forceinline__ v8f wmma_f16(const v16h& a, const v16h& b, const v8f& c) {
  return __builtin_amdgcn_wmma_f32_16x16x32_f16(false, a, false, b, (short)0, c, false, false);
}

// ---------------------------------------------------------------------------
__global__ __launch_bounds__(256) void hashpinn_fused(
    const float* __restrict__ x,  const float* __restrict__ table,
    const float* __restrict__ W1, const float* __restrict__ b1,
    const float* __restrict__ W2, const float* __restrict__ b2,
    const float* __restrict__ W3, const float* __restrict__ b3,
    const float* __restrict__ Wo, const float* __restrict__ bo,
    float* __restrict__ out, int N)
{
  __shared__ __align__(16) _Float16 s_feat[256][32];      // 16 KB per-point features
  __shared__ __align__(16) _Float16 s_h[8][16][HSTRIDE];  // 18 KB per-wave hidden tile
  __shared__ __align__(16) _Float16 s_W1t[64 * 32];       //  4 KB  W1^T  [n][k]
  __shared__ __align__(16) _Float16 s_W2t[64 * 64];       //  8 KB  W2^T  [n][k]
  __shared__ __align__(16) _Float16 s_W3t[64 * 64];       //  8 KB  W3^T  [n][k]
  __shared__ __align__(16) _Float16 s_Wo[64];
  __shared__ float s_b1[64], s_b2[64], s_b3[64];

  const int tid = threadIdx.x;

  // ---- stage weights fp32 -> fp16, transposed to [n][k], into LDS ----
  for (int i = tid; i < 32 * 64; i += 256) {               // W1 is [k=32][n=64]
    const int k = i >> 6, n = i & 63;
    s_W1t[n * 32 + k] = (_Float16)W1[i];
  }
  for (int i = tid; i < 64 * 64; i += 256) {               // W2/W3 are [k=64][n=64]
    const int k = i >> 6, n = i & 63;
    s_W2t[n * 64 + k] = (_Float16)W2[i];
    s_W3t[n * 64 + k] = (_Float16)W3[i];
  }
  if (tid < 64) {
    s_Wo[tid] = (_Float16)Wo[tid];
    s_b1[tid] = b1[tid]; s_b2[tid] = b2[tid]; s_b3[tid] = b3[tid];
  }
  const float bo_v = bo[0];

  // ---- phase 1: hash-grid encode (one point per thread) ----
  const int pidx = blockIdx.x * 256 + tid;
  if (pidx < N) {
    const float px = x[3 * pidx + 0];
    const float py = x[3 * pidx + 1];
    const float pz = x[3 * pidx + 2];
#pragma unroll
    for (int l = 0; l < NLEV; ++l) {
      const int res = RES[l];
      const float rf = (float)(res - 1);
      const float gx = px * rf, gy = py * rf, gz = pz * rf;
      int ix = (int)floorf(gx); ix = ix < 0 ? 0 : (ix > res - 2 ? res - 2 : ix);
      int iy = (int)floorf(gy); iy = iy < 0 ? 0 : (iy > res - 2 ? res - 2 : iy);
      int iz = (int)floorf(gz); iz = iz < 0 ? 0 : (iz > res - 2 ? res - 2 : iz);
      const float fx = gx - (float)ix, fy = gy - (float)iy, fz = gz - (float)iz;
      const float wx[2] = {1.f - fx, fx};
      const float wy[2] = {1.f - fy, fy};
      const float wz[2] = {1.f - fz, fz};
      const float* tl = table + (size_t)l * (size_t)(TBL_SZ * 2);
      float f0 = 0.f, f1 = 0.f;
      const bool dense = ((long long)res * res * res) <= (long long)TBL_SZ;  // folds per-level
#pragma unroll
      for (int c = 0; c < 8; ++c) {
        const int ox = (c >> 2) & 1, oy = (c >> 1) & 1, oz = c & 1;
        const int cx = ix + ox, cy = iy + oy, cz = iz + oz;
        uint32_t id;
        if (dense) {
          id = (uint32_t)(cx + res * (cy + res * cz));
        } else {
          id = (((uint32_t)cx * 1u) ^ ((uint32_t)cy * 2654435761u) ^
                ((uint32_t)cz * 805459861u)) & (uint32_t)(TBL_SZ - 1);
        }
        const float2 tv = *reinterpret_cast<const float2*>(tl + 2u * id);
        const float w = wx[ox] * wy[oy] * wz[oz];
        f0 += w * tv.x;
        f1 += w * tv.y;
      }
      s_feat[tid][2 * l]     = (_Float16)f0;
      s_feat[tid][2 * l + 1] = (_Float16)f1;
    }
  } else {
#pragma unroll
    for (int l = 0; l < 2 * NLEV; ++l) s_feat[tid][l] = (_Float16)0.f;
  }
  __syncthreads();  // weights + features visible block-wide

  // ---- phase 2: per-wave WMMA MLP; wave w owns points 32w..32w+31 = 2 tiles ----
  const int wave = tid >> 5, lane = tid & 31;
  _Float16* h = &s_h[wave][0][0];
  const int blockBase = blockIdx.x * 256;

#pragma unroll 1
  for (int t = 0; t < 2; ++t) {
    const int tile = wave * 2 + t;

    // ---- layer 1: [16x32] @ [32x64] + b1, ReLU ----
    {
      const v16h A = load_A16x32(&s_feat[tile * 16][0], 32, lane);
      v8f acc[4];
#pragma unroll
      for (int jg = 0; jg < 4; ++jg) {
        const v16h B = load_Bt32x16(s_W1t, 32, jg * 16, 0, lane);
        const v8f c0 = {};
        acc[jg] = wmma_f16(A, B, c0);
      }
#pragma unroll
      for (int jg = 0; jg < 4; ++jg) store_relu(h, acc[jg], s_b1, jg * 16, lane);
      asm volatile("s_wait_dscnt 0x0" ::: "memory");  // act stores -> frag loads
    }

    // ---- layers 2 & 3: [16x64] @ [64x64] + b, ReLU (K split 32+32) ----
#pragma unroll 1
    for (int ly = 0; ly < 2; ++ly) {
      const _Float16* Wt = (ly == 0) ? (const _Float16*)s_W2t : (const _Float16*)s_W3t;
      const float*    bb = (ly == 0) ? (const float*)s_b2     : (const float*)s_b3;

      const v16h A0 = load_A16x32(h,      HSTRIDE, lane);   // K = 0..31
      const v16h A1 = load_A16x32(h + 32, HSTRIDE, lane);   // K = 32..63
      v8f acc[4];
#pragma unroll
      for (int jg = 0; jg < 4; ++jg) {
        v8f c0 = {};
        c0      = wmma_f16(A0, load_Bt32x16(Wt, 64, jg * 16, 0,  lane), c0);
        acc[jg] = wmma_f16(A1, load_Bt32x16(Wt, 64, jg * 16, 32, lane), c0);
      }
#pragma unroll
      for (int jg = 0; jg < 4; ++jg) store_relu(h, acc[jg], bb, jg * 16, lane);
      asm volatile("s_wait_dscnt 0x0" ::: "memory");
    }

    // ---- output layer: [16x64] @ [64x1]; Wo packed into B column 0 ----
    {
      const v16h A0 = load_A16x32(h,      HSTRIDE, lane);
      const v16h A1 = load_A16x32(h + 32, HSTRIDE, lane);
      const int n = lane & 15, kb = (lane >> 4) << 4;
      const v8h w0lo = *reinterpret_cast<const v8h*>(s_Wo + kb);
      const v8h w0hi = *reinterpret_cast<const v8h*>(s_Wo + kb + 8);
      const v8h w1lo = *reinterpret_cast<const v8h*>(s_Wo + 32 + kb);
      const v8h w1hi = *reinterpret_cast<const v8h*>(s_Wo + 32 + kb + 8);
      v16h B0, B1;
#pragma unroll
      for (int e = 0; e < 8; ++e) {
        B0[e]     = (n == 0) ? w0lo[e] : (_Float16)0.f;
        B0[8 + e] = (n == 0) ? w0hi[e] : (_Float16)0.f;
        B1[e]     = (n == 0) ? w1lo[e] : (_Float16)0.f;
        B1[8 + e] = (n == 0) ? w1hi[e] : (_Float16)0.f;
      }
      v8f c0 = {};
      c0 = wmma_f16(A0, B0, c0);
      const v8f acc = wmma_f16(A1, B1, c0);
      if (n == 0) {  // lanes 0 and 16 hold D column 0: points M = r + 8*half
        const int half = lane >> 4;
#pragma unroll
        for (int r = 0; r < 8; ++r) {
          const int p = blockBase + tile * 16 + r + half * 8;
          if (p < N) out[p] = acc[r] + bo_v;
        }
      }
    }
  }
}

// ---------------------------------------------------------------------------
extern "C" void kernel_launch(void* const* d_in, const int* in_sizes, int n_in,
                              void* d_out, int out_size, void* d_ws, size_t ws_size,
                              hipStream_t stream) {
  const float* x  = (const float*)d_in[0];
  const float* tb = (const float*)d_in[1];
  const float* W1 = (const float*)d_in[2];
  const float* b1 = (const float*)d_in[3];
  const float* W2 = (const float*)d_in[4];
  const float* b2 = (const float*)d_in[5];
  const float* W3 = (const float*)d_in[6];
  const float* b3 = (const float*)d_in[7];
  const float* Wo = (const float*)d_in[8];
  const float* bo = (const float*)d_in[9];
  float* out = (float*)d_out;

  const int N = in_sizes[0] / 3;           // x is [N,3]
  const int blocks = (N + 255) / 256;
  hipLaunchKernelGGL(hashpinn_fused, dim3(blocks), dim3(256), 0, stream,
                     x, tb, W1, b1, W2, b2, W3, b3, Wo, bo, out, N);
}
